// HierarchicalInductiveAutoencoder_7902739825034
// MI455X (gfx1250) — matrix operand order repaired and checked by reading, not compiled, not run on hardware
//
#include <hip/hip_runtime.h>
#include <hip/hip_bf16.h>
#include <math.h>

typedef __attribute__((ext_vector_type(16))) _Float16 v16h;
typedef __attribute__((ext_vector_type(8)))  float    v8f;
typedef __attribute__((ext_vector_type(4)))  float    v4f;

#define ACT_NONE 0
#define ACT_RELU 1
#define ACT_TANH 2

// ---------------------------------------------------------------------------
// Generic GEMM: C(M,N) = act(A(M,K) @ W(K,N) + bias)
// REQUIRES: M % 16 == 0, N % 16 == 0, A rows 16B-aligned (K % 4 == 0).
// One 16x16 output tile per wave32, K stepped by 32 via v_wmma_f32_16x16x32_f16.
// A/B/C lane layouts per CDNA5 ISA 7.12.2 (wave32). Main loop is branch-free:
// A tile = 4x b128 loads/lane, B tile = 16x b32 loads/lane, + prefetch of the
// next K chunk. K-tail uses clamped addresses + selects (no exec divergence).
// ---------------------------------------------------------------------------
__global__ void gemm_bias_act(const float* __restrict__ A,
                              const float* __restrict__ Wm,
                              const float* __restrict__ bias,
                              float* __restrict__ C,
                              int M, int N, int K, int act) {
  const int lane = threadIdx.x & 31;
  const int half = lane >> 4;       // 0: lanes 0-15, 1: lanes 16-31
  const int mloc = lane & 15;
  const int tm = blockIdx.y, tn = blockIdx.x;
  const int row  = tm * 16 + mloc;  // A row (same row set in both wave halves)
  const int coln = tn * 16 + mloc;  // B/C column

  const float* Arow = A + (size_t)row * K;
  const float* Wcol = Wm + coln;

  v8f acc = {};
  const int kfull = K & ~31;

  for (int kk = 0; kk < kfull; kk += 32) {
    // ---- A tile: lanes need K = kk+half*8+{0..7} and kk+16+half*8+{0..7} ----
    const float* ab = Arow + kk + half * 8;
    v4f a0 = *(const v4f*)(ab);
    v4f a1 = *(const v4f*)(ab + 4);
    v4f a2 = *(const v4f*)(ab + 16);
    v4f a3 = *(const v4f*)(ab + 20);
    v16h a;
#pragma unroll
    for (int t = 0; t < 4; ++t) {
      a[t]      = (_Float16)a0[t];
      a[4 + t]  = (_Float16)a1[t];
      a[8 + t]  = (_Float16)a2[t];
      a[12 + t] = (_Float16)a3[t];
    }
    // ---- B tile: rows kk+half*8+{0..7} and kk+16+half*8+{0..7} at fixed column, stride N ----
    const float* wp = Wcol + (size_t)(kk + half * 8) * N;
    v16h b;
#pragma unroll
    for (int t = 0; t < 16; ++t) b[t] = (_Float16)wp[(size_t)(t < 8 ? t : t + 8) * N];

    if (kk + 32 < kfull) {  // prefetch next chunk (global_prefetch_b8)
      __builtin_prefetch(Arow + kk + 32 + half * 8, 0, 1);
      __builtin_prefetch(Wcol + (size_t)(kk + 32 + half * 16) * N, 0, 1);
    }

    acc = __builtin_amdgcn_wmma_f32_16x16x32_f16(false, a, false, b,
                                                 (short)0, acc, false, false);
  }

  // ---- K tail (K % 32 != 0): clamped addresses + selects, no divergence ----
  if (kfull < K) {
    v16h a, b;
#pragma unroll
    for (int t = 0; t < 16; ++t) {
      int k = kfull + half * 8 + (t < 8 ? t : t + 8);
      int ks = (k < K) ? k : (K - 1);
      float x = Arow[ks];
      a[t] = (k < K) ? (_Float16)x : (_Float16)0.0f;
    }
#pragma unroll
    for (int t = 0; t < 16; ++t) {
      int k = kfull + half * 8 + (t < 8 ? t : t + 8);
      int ks = (k < K) ? k : (K - 1);
      float y = Wcol[(size_t)ks * N];
      b[t] = (k < K) ? (_Float16)y : (_Float16)0.0f;
    }
    acc = __builtin_amdgcn_wmma_f32_16x16x32_f16(false, a, false, b,
                                                 (short)0, acc, false, false);
  }

  // ---- epilogue: bias + activation; VGPR v holds M=v / v+8 per wave half ----
  const float bv = bias[coln];
#pragma unroll
  for (int v = 0; v < 8; ++v) {
    int m = tm * 16 + v + 8 * half;
    float val = acc[v] + bv;
    if (act == ACT_RELU)      val = fmaxf(val, 0.0f);
    else if (act == ACT_TANH) val = tanhf(val);
    C[(size_t)m * N + coln] = val;
  }
}

// ---------------------------------------------------------------------------
// Multi-resolution hash encode + spatial mean (coords are batch-invariant).
// One block per level; deterministic LDS tree reduction.
// ---------------------------------------------------------------------------
__global__ void hash_pool_mean(const float* __restrict__ tables, int table_stride,
                               int num_levels, int fpl, float basef, float finestf,
                               int log2size, int H, int W, float* __restrict__ out) {
  const int lvl = blockIdx.x;
  double gl = (log((double)finestf) - log((double)basef)) / (double)(num_levels - 1);
  int res = (int)floor((double)basef * exp(gl * (double)lvl));
  long long r3 = (long long)res * res * res;
  long long cap = 1LL << log2size;
  unsigned size = (unsigned)(r3 < cap ? r3 : cap);
  const float* tab = tables + (size_t)lvl * table_stride;

  float sums[8];
  for (int c = 0; c < 8; ++c) sums[c] = 0.0f;

  const int total = H * W;
  for (int p = threadIdx.x; p < total; p += blockDim.x) {
    int i = p / W, j = p % W;
    float x = (float)j / (float)(W - 1);
    float y = (float)i / (float)(H - 1);
    int gx = (int)floorf(x * (float)res);
    int gy = (int)floorf(y * (float)res);
    int gz = (int)floorf(0.5f * (float)res);
    unsigned h = (unsigned)gx * 73856093u ^ (unsigned)gy * 19349663u ^
                 (unsigned)gz * 83492791u;
    int hi = (int)h;
    unsigned uabs = (hi < 0) ? (unsigned)(-(long long)hi) : (unsigned)hi;
    unsigned idx = uabs % size;
    const float* f = tab + (size_t)idx * fpl;
    for (int c = 0; c < fpl; ++c) sums[c] += f[c];
  }

  __shared__ float sh[256];
  for (int c = 0; c < fpl; ++c) {
    sh[threadIdx.x] = sums[c];
    __syncthreads();
    for (int s = 128; s > 0; s >>= 1) {
      if ((int)threadIdx.x < s) sh[threadIdx.x] += sh[threadIdx.x + s];
      __syncthreads();
    }
    if (threadIdx.x == 0) out[lvl * fpl + c] = sh[0] / (float)total;
    __syncthreads();
  }
}

// ---------------------------------------------------------------------------
// ToF 3D conv (VALID) + bias + relu:  (B,4,4,100,1) * (3,3,5,1,16) -> (B,2,2,96,16)
// ---------------------------------------------------------------------------
__global__ void conv3d_tof(const float* __restrict__ in, const float* __restrict__ K,
                           const float* __restrict__ bias, float* __restrict__ out,
                           int total) {
  int idx = blockIdx.x * blockDim.x + threadIdx.x;
  if (idx >= total) return;
  int oc = idx % 16;
  int w  = (idx / 16) % 96;
  int h2 = (idx / (16 * 96)) % 2;
  int d  = (idx / (16 * 96 * 2)) % 2;
  int b  = idx / (16 * 96 * 2 * 2);
  float s = bias[oc];
  for (int kd = 0; kd < 3; ++kd)
    for (int kh = 0; kh < 3; ++kh)
      for (int kw = 0; kw < 5; ++kw) {
        float xv = in[(((b * 4 + d + kd) * 4 + h2 + kh) * 100 + w + kw)];
        s += xv * K[((kd * 3 + kh) * 5 + kw) * 16 + oc];
      }
  out[idx] = fmaxf(s, 0.0f);
}

// ---------------------------------------------------------------------------
// ToF 1D conv (VALID) + bias + relu on rs=(B,64,96): (5,96,32) -> (B,60,32)
// rs[b,i,j] = mp_flat[b*6144 + i*96 + j]
// ---------------------------------------------------------------------------
__global__ void conv1d_tof(const float* __restrict__ mp, const float* __restrict__ K,
                           const float* __restrict__ bias, float* __restrict__ out,
                           int total) {
  int idx = blockIdx.x * blockDim.x + threadIdx.x;
  if (idx >= total) return;
  int oc = idx % 32;
  int ow = (idx / 32) % 60;
  int b  = idx / (32 * 60);
  float s = bias[oc];
  for (int kw = 0; kw < 5; ++kw)
    for (int c = 0; c < 96; ++c)
      s += mp[b * 6144 + (ow + kw) * 96 + c] * K[(kw * 96 + c) * 32 + oc];
  out[idx] = fmaxf(s, 0.0f);
}

// ---------------------------------------------------------------------------
// Fused 3x3 VALID conv + bias + relu + 2x2 maxpool (VALID)
// in: (B,H,W,Cin) -> out: (B,OH,OW,Cout), OH=(H-2)/2 etc.
// ---------------------------------------------------------------------------
__global__ void conv3x3_relu_pool(const float* __restrict__ in, int H, int Wd, int Cin,
                                  const float* __restrict__ K, const float* __restrict__ bias,
                                  int Cout, float* __restrict__ out, int OH, int OW,
                                  int total) {
  int idx = blockIdx.x * blockDim.x + threadIdx.x;
  if (idx >= total) return;
  int oc = idx % Cout;
  int j  = (idx / Cout) % OW;
  int i  = (idx / (Cout * OW)) % OH;
  int b  = idx / (Cout * OW * OH);
  float bv = bias[oc];
  float best = -1e30f;
  for (int pi = 0; pi < 2; ++pi)
    for (int pj = 0; pj < 2; ++pj) {
      int r0 = 2 * i + pi, c0 = 2 * j + pj;
      float s = bv;
      for (int kh = 0; kh < 3; ++kh)
        for (int kw = 0; kw < 3; ++kw) {
          const float* ip = in + (((size_t)(b * H + r0 + kh) * Wd + c0 + kw) * Cin);
          const float* kp = K + ((kh * 3 + kw) * Cin) * Cout + oc;
          for (int ic = 0; ic < Cin; ++ic) s += ip[ic] * kp[ic * Cout];
        }
      s = fmaxf(s, 0.0f);
      best = fmaxf(best, s);
    }
  out[idx] = best;
}

// ---------------------------------------------------------------------------
// 3x3 VALID conv + bias + relu + global spatial mean.
// in: (B,62,62,64) -> cnn: (B,128).  One block per (b,oc).
// ---------------------------------------------------------------------------
__global__ void conv3_mean(const float* __restrict__ in, const float* __restrict__ K,
                           const float* __restrict__ bias, float* __restrict__ out) {
  int b = blockIdx.x >> 7;
  int oc = blockIdx.x & 127;
  float bv = bias[oc];
  float acc = 0.0f;
  for (int pos = threadIdx.x; pos < 3600; pos += blockDim.x) {
    int i = pos / 60, j = pos % 60;
    float s = bv;
    for (int kh = 0; kh < 3; ++kh)
      for (int kw = 0; kw < 3; ++kw) {
        const float* ip = in + (((size_t)(b * 62 + i + kh) * 62 + j + kw) * 64);
        const float* kp = K + ((kh * 3 + kw) * 64) * 128 + oc;
        for (int ic = 0; ic < 64; ++ic) s += ip[ic] * kp[ic * 128];
      }
    acc += fmaxf(s, 0.0f);
  }
  __shared__ float sh[256];
  sh[threadIdx.x] = acc;
  __syncthreads();
  for (int s = 128; s > 0; s >>= 1) {
    if ((int)threadIdx.x < s) sh[threadIdx.x] += sh[threadIdx.x + s];
    __syncthreads();
  }
  if (threadIdx.x == 0) out[b * 128 + oc] = sh[0] / 3600.0f;
}

// ---------------------------------------------------------------------------
// conv_transpose 3x3 stride 2 'SAME' (JAX pad_a=2, pad_b=1, unflipped kernel),
// + bias + relu.  in: (B,IH,IW,Cin) -> out: (B,2*IH,2*IW,Cout)
// ---------------------------------------------------------------------------
__global__ void convT3x3_s2_relu(const float* __restrict__ in, int IH, int IW, int Cin,
                                 const float* __restrict__ K, const float* __restrict__ bias,
                                 int Cout, float* __restrict__ out, int OH, int OW,
                                 int total) {
  int idx = blockIdx.x * blockDim.x + threadIdx.x;
  if (idx >= total) return;
  int oc = idx % Cout;
  int ow = (idx / Cout) % OW;
  int oh = (idx / (Cout * OW)) % OH;
  int b  = idx / (Cout * OW * OH);
  float s = bias[oc];
  for (int kh = 0; kh < 3; ++kh) {
    int dh = oh + kh - 2;
    if (dh < 0 || (dh & 1)) continue;
    int ih = dh >> 1;
    if (ih >= IH) continue;
    for (int kw = 0; kw < 3; ++kw) {
      int dw = ow + kw - 2;
      if (dw < 0 || (dw & 1)) continue;
      int iw = dw >> 1;
      if (iw >= IW) continue;
      const float* ip = in + (((size_t)(b * IH + ih) * IW + iw) * Cin);
      const float* kp = K + ((kh * 3 + kw) * Cin) * Cout + oc;
      for (int ic = 0; ic < Cin; ++ic) s += ip[ic] * kp[ic * Cout];
    }
  }
  out[idx] = fmaxf(s, 0.0f);
}

// ---------------------------------------------------------------------------
// 3x3 SAME conv (Cin=32, Cout=1) + bias + sigmoid -> depth output
// ---------------------------------------------------------------------------
__global__ void conv3x3_same_sig(const float* __restrict__ in, int H, int Wd,
                                 const float* __restrict__ K, const float* __restrict__ bias,
                                 float* __restrict__ out, int total) {
  int idx = blockIdx.x * blockDim.x + threadIdx.x;
  if (idx >= total) return;
  int j = idx % Wd;
  int i = (idx / Wd) % H;
  int b = idx / (Wd * H);
  float s = bias[0];
  for (int kh = 0; kh < 3; ++kh) {
    int ii = i + kh - 1;
    if (ii < 0 || ii >= H) continue;
    for (int kw = 0; kw < 3; ++kw) {
      int jj = j + kw - 1;
      if (jj < 0 || jj >= Wd) continue;
      const float* ip = in + (((size_t)(b * H + ii) * Wd + jj) * 32);
      const float* kp = K + (kh * 3 + kw) * 32;
      for (int ic = 0; ic < 32; ++ic) s += ip[ic] * kp[ic];
    }
  }
  out[idx] = 1.0f / (1.0f + expf(-s));
}

// ---------------------------------------------------------------------------
// 2D strided copy; srcStride==0 broadcasts one source row to all rows.
// ---------------------------------------------------------------------------
__global__ void copy2d(float* __restrict__ dst, int dstStride,
                       const float* __restrict__ src, int srcStride,
                       int rows, int cols) {
  int idx = blockIdx.x * blockDim.x + threadIdx.x;
  int total = rows * cols;
  if (idx >= total) return;
  int r = idx / cols, c = idx % cols;
  dst[(size_t)r * dstStride + c] = src[(size_t)r * srcStride + c];
}

// ---------------------------------------------------------------------------

static inline int cdiv(int a, int b) { return (a + b - 1) / b; }

extern "C" void kernel_launch(void* const* d_in, const int* in_sizes, int n_in,
                              void* d_out, int out_size, void* d_ws, size_t ws_size,
                              hipStream_t stream) {
  const int B = 16;
  const float* tof        = (const float*)d_in[0];
  const float* visual     = (const float*)d_in[1];
  const float* vis_tables = (const float*)d_in[2];
  const float* dec_tables = (const float*)d_in[3];
  const float* W_ga = (const float*)d_in[4];   const float* b_ga = (const float*)d_in[5];
  const float* W_pd = (const float*)d_in[6];   const float* b_pd = (const float*)d_in[7];
  const float* K3d  = (const float*)d_in[8];   const float* b3d  = (const float*)d_in[9];
  const float* K1d  = (const float*)d_in[10];  const float* b1d  = (const float*)d_in[11];
  const float* W_phys = (const float*)d_in[12];const float* b_phys = (const float*)d_in[13];
  const float* Kc1 = (const float*)d_in[14];   const float* bc1 = (const float*)d_in[15];
  const float* Kc2 = (const float*)d_in[16];   const float* bc2 = (const float*)d_in[17];
  const float* Kc3 = (const float*)d_in[18];   const float* bc3 = (const float*)d_in[19];
  const float* W_vis = (const float*)d_in[20]; const float* b_vis = (const float*)d_in[21];
  const float* W_modal = (const float*)d_in[22]; const float* b_modal = (const float*)d_in[23];
  const float* W_h0 = (const float*)d_in[24];  const float* b_h0 = (const float*)d_in[25];
  const float* W_h1 = (const float*)d_in[26];  const float* b_h1 = (const float*)d_in[27];
  const float* W_h2 = (const float*)d_in[28];  const float* b_h2 = (const float*)d_in[29];
  const float* W_hier = (const float*)d_in[30];const float* b_hier = (const float*)d_in[31];
  const float* W_d1 = (const float*)d_in[32];  const float* b_d1 = (const float*)d_in[33];
  const float* W_d2 = (const float*)d_in[34];  const float* b_d2 = (const float*)d_in[35];
  const float* Kt1 = (const float*)d_in[36];   const float* bt1 = (const float*)d_in[37];
  const float* Kt2 = (const float*)d_in[38];   const float* bt2 = (const float*)d_in[39];
  const float* Kdf = (const float*)d_in[40];   const float* bdf = (const float*)d_in[41];
  const float* W_p1 = (const float*)d_in[42];  const float* b_p1 = (const float*)d_in[43];
  const float* W_p2 = (const float*)d_in[44];  const float* b_p2 = (const float*)d_in[45];

  float* ws = (float*)d_ws;
  // workspace layout (floats); BIG1 aliases {p1, t2}; BIG2 aliases {p2, t1}
  const size_t OFF_MP     = 0;                      // 98304
  const size_t OFF_TMP    = OFF_MP + 98304;         // 30720
  const size_t OFF_GAU    = OFF_TMP + 30720;        // 16384
  const size_t OFF_PEAKS  = OFF_GAU + 16384;        // 8192
  const size_t OFF_COMB   = OFF_PEAKS + 8192;       // 38912
  const size_t OFF_TOFENC = OFF_COMB + 38912;       // 2048
  const size_t OFF_H48    = OFF_TOFENC + 2048;      // 64 (48 used)
  const size_t OFF_CNN    = OFF_H48 + 64;           // 2048
  const size_t OFF_VISCAT = OFF_CNN + 2048;         // 2816
  const size_t OFF_VISENC = OFF_VISCAT + 2816;      // 2048
  const size_t OFF_FCAT   = OFF_VISENC + 2048;      // 4096
  const size_t OFF_FUSED  = OFF_FCAT + 4096;        // 4096
  const size_t OFF_H0     = OFF_FUSED + 4096;       // 4096
  const size_t OFF_H1     = OFF_H0 + 4096;          // 2048
  const size_t OFF_H2     = OFF_H1 + 2048;          // 1024
  const size_t OFF_HCAT   = OFF_H2 + 1024;          // 7168
  const size_t OFF_FINAL  = OFF_HCAT + 7168;        // 4096
  const size_t OFF_DEC64  = OFF_FINAL + 4096;       // 64
  const size_t OFF_POOLED = OFF_DEC64 + 64;         // 5120
  const size_t OFF_D1     = OFF_POOLED + 5120;      // 8192
  const size_t OFF_D2     = OFF_D1 + 8192;          // 16384
  const size_t OFF_Q1     = OFF_D2 + 16384;         // 8192
  const size_t OFF_BIG2   = OFF_Q1 + 8192;          // 4,194,304 (p2 / t1)
  const size_t OFF_BIG1   = OFF_BIG2 + 4194304;     // 8,388,608 (p1 / t2)

  float* mp     = ws + OFF_MP;
  float* tmp    = ws + OFF_TMP;
  float* gau    = ws + OFF_GAU;
  float* peaks  = ws + OFF_PEAKS;
  float* comb   = ws + OFF_COMB;
  float* tofenc = ws + OFF_TOFENC;
  float* h48    = ws + OFF_H48;
  float* cnn    = ws + OFF_CNN;
  float* viscat = ws + OFF_VISCAT;
  float* visenc = ws + OFF_VISENC;
  float* fcat   = ws + OFF_FCAT;
  float* fused  = ws + OFF_FUSED;
  float* h0     = ws + OFF_H0;
  float* h1     = ws + OFF_H1;
  float* h2     = ws + OFF_H2;
  float* hcat   = ws + OFF_HCAT;
  float* finalb = ws + OFF_FINAL;
  float* dec64  = ws + OFF_DEC64;
  float* pooled = ws + OFF_POOLED;
  float* dd1    = ws + OFF_D1;
  float* dd2    = ws + OFF_D2;
  float* q1     = ws + OFF_Q1;
  float* p1     = ws + OFF_BIG1;   // (16,127,127,32)
  float* p2     = ws + OFF_BIG2;   // (16,62,62,64)
  float* t1     = ws + OFF_BIG2;   // (16,64,64,64)  aliases p2 (p2 dead by then)
  float* t2     = ws + OFF_BIG1;   // (16,128,128,32) aliases p1

  float* depth = (float*)d_out;                 // (16,128,128,1)
  float* pose  = (float*)d_out + 262144;        // (16,32,32,18)

  const dim3 wave(32);
#define GEMM(A_, W_, b_, C_, M_, N_, K_, act_) \
  gemm_bias_act<<<dim3(cdiv((N_),16), cdiv((M_),16)), wave, 0, stream>>>((A_),(W_),(b_),(C_),(M_),(N_),(K_),(act_))

  // ---------------- ToF encoder ----------------
  conv3d_tof<<<cdiv(98304,256),256,0,stream>>>(tof, K3d, b3d, mp, 98304);
  conv1d_tof<<<cdiv(30720,256),256,0,stream>>>(mp, K1d, b1d, tmp, 30720);
  GEMM(tof,  W_ga, b_ga, gau,   256, 64, 100, ACT_RELU);   // flat_h (256,100)
  GEMM(gau,  W_pd, b_pd, peaks, 256, 32,  64, ACT_RELU);
  copy2d<<<cdiv(16*512,256),256,0,stream>>>(comb,        2432, peaks, 512, 16, 512);
  copy2d<<<cdiv(16*1920,256),256,0,stream>>>(comb + 512, 2432, tmp,  1920, 16, 1920);
  GEMM(comb, W_phys, b_phys, tofenc, 16, 128, 2432, ACT_TANH);

  // ---------------- Visual encoder ----------------
  hash_pool_mean<<<12,256,0,stream>>>(vis_tables, (1<<19)*4, 12, 4, 8.0f, 256.0f, 19,
                                      256, 256, h48);
  {
    int tot1 = B*127*127*32;
    conv3x3_relu_pool<<<cdiv(tot1,256),256,0,stream>>>(visual, 256, 256, 3, Kc1, bc1, 32,
                                                       p1, 127, 127, tot1);
    int tot2 = B*62*62*64;
    conv3x3_relu_pool<<<cdiv(tot2,256),256,0,stream>>>(p1, 127, 127, 32, Kc2, bc2, 64,
                                                       p2, 62, 62, tot2);
  }
  conv3_mean<<<B*128,256,0,stream>>>(p2, Kc3, bc3, cnn);
  copy2d<<<cdiv(16*128,256),256,0,stream>>>(viscat,       176, cnn, 128, 16, 128);
  copy2d<<<cdiv(16*48,256),256,0,stream>>>(viscat + 128,  176, h48,   0, 16, 48);
  GEMM(viscat, W_vis, b_vis, visenc, 16, 128, 176, ACT_TANH);

  // ---------------- Fusion + hierarchy ----------------
  copy2d<<<cdiv(16*128,256),256,0,stream>>>(fcat,       256, tofenc, 128, 16, 128);
  copy2d<<<cdiv(16*128,256),256,0,stream>>>(fcat + 128, 256, visenc, 128, 16, 128);
  GEMM(fcat,  W_modal, b_modal, fused, 16, 256, 256, ACT_RELU);
  GEMM(fused, W_h0,    b_h0,    h0,    16, 256, 256, ACT_RELU);
  GEMM(h0,    W_h1,    b_h1,    h1,    16, 128, 256, ACT_RELU);
  GEMM(h1,    W_h2,    b_h2,    h2,    16,  64, 128, ACT_RELU);
  copy2d<<<cdiv(16*256,256),256,0,stream>>>(hcat,       448, h0, 256, 16, 256);
  copy2d<<<cdiv(16*128,256),256,0,stream>>>(hcat + 256, 448, h1, 128, 16, 128);
  copy2d<<<cdiv(16*64,256),256,0,stream>>>(hcat + 384,  448, h2,  64, 16, 64);
  GEMM(hcat, W_hier, b_hier, finalb, 16, 256, 448, ACT_RELU);

  // ---------------- Decoder: pooled = [mean(dec_hash) ⊕ final] ----------------
  hash_pool_mean<<<8,256,0,stream>>>(dec_tables, 32768*8, 8, 8, 4.0f, 32.0f, 19,
                                     32, 32, dec64);
  copy2d<<<cdiv(16*64,256),256,0,stream>>>(pooled,      320, dec64,    0, 16, 64);
  copy2d<<<cdiv(16*256,256),256,0,stream>>>(pooled + 64,320, finalb, 256, 16, 256);

  // ---------------- Depth head ----------------
  GEMM(pooled, W_d1, b_d1, dd1, 16,  512, 320, ACT_RELU);
  GEMM(dd1,    W_d2, b_d2, dd2, 16, 1024, 512, ACT_RELU);   // -> (B,32,32,1)
  {
    int tot = B*64*64*64;
    convT3x3_s2_relu<<<cdiv(tot,256),256,0,stream>>>(dd2, 32, 32, 1, Kt1, bt1, 64,
                                                     t1, 64, 64, tot);
    int tot2 = B*128*128*32;
    convT3x3_s2_relu<<<cdiv(tot2,256),256,0,stream>>>(t1, 64, 64, 64, Kt2, bt2, 32,
                                                      t2, 128, 128, tot2);
    int totd = B*128*128;
    conv3x3_same_sig<<<cdiv(totd,256),256,0,stream>>>(t2, 128, 128, Kdf, bdf, depth, totd);
  }

  // ---------------- Pose head (writes d_out directly) ----------------
  GEMM(pooled, W_p1, b_p1, q1, 16, 512, 320, ACT_RELU);
  GEMM(q1, W_p2, b_p2, pose, 16, 18432, 512, ACT_RELU);

#undef GEMM
  (void)in_sizes; (void)n_in; (void)out_size; (void)ws_size;
}
